// KanLayer_original_45079976739123
// MI455X (gfx1250) — matrix operand order repaired and checked by reading, not compiled, not run on hardware
//
#include <hip/hip_runtime.h>
#include <hip/hip_bf16.h>
#include <math.h>

// ---------------- problem constants ----------------
#define BATCH    8192
#define IN_DIM   256
#define OUT_DIM  64
#define N_INT    32
#define NCOEF    (N_INT + 3)          // 35
#define KSPLINE  (IN_DIM * NCOEF)     // 8960
#define KTOTAL   (KSPLINE + IN_DIM)   // 9216 (base/silu rows fused into GEMM K)
#define KCHUNK   32                   // WMMA K per instruction (f16)
#define NKC      (KTOTAL / KCHUNK)    // 288 K-chunks
#define NTILES   (OUT_DIM / 16)       // 4 N-tiles
#define SCK      1536                 // K per LDS super-chunk (48KB A panel)
#define NSC      (KTOTAL / SCK)       // 6 super-chunks
#define CPS      (SCK / KCHUNK)       // 48 chunks per super-chunk
#define BCSTRIDE (NTILES * 32 * 16)   // f16 elems between consecutive K-chunk frags
#define EPSV     1e-5f
#define NORM_DIV 3.0f

// B fragment table: NKC chunks * NTILES ntiles * 32 lanes * 16 f16 each
#define BFRAG_ELEMS (NKC * NTILES * 32 * 16)   // 589824 f16 = 1.125 MB

typedef __attribute__((ext_vector_type(16))) _Float16 v16h;
typedef __attribute__((ext_vector_type(8)))  float    v8f;

// ---------------- kernel 1: batchnorm stats ----------------
// One block per column d: mean[d] and scale[d] = 1/(sqrt(var+eps)*3).
__global__ __launch_bounds__(256)
void kan_bn_stats(const float* __restrict__ x,
                  float* __restrict__ mean, float* __restrict__ scale) {
  const int d = blockIdx.x;
  const int tid = threadIdx.x;
  float s = 0.f, s2 = 0.f;
  for (int b = tid; b < BATCH; b += 256) {
    float v = x[b * IN_DIM + d];
    s += v; s2 += v * v;
  }
  __shared__ float sh[256];
  __shared__ float sh2[256];
  sh[tid] = s; sh2[tid] = s2;
  __syncthreads();
  for (int off = 128; off > 0; off >>= 1) {
    if (tid < off) { sh[tid] += sh[tid + off]; sh2[tid] += sh2[tid + off]; }
    __syncthreads();
  }
  if (tid == 0) {
    float m = sh[0] * (1.0f / BATCH);
    float var = sh2[0] * (1.0f / BATCH) - m * m;
    mean[d] = m;
    scale[d] = 1.0f / (sqrtf(var + EPSV) * NORM_DIV);
  }
}

// ---------------- kernel 2: pack B into WMMA fragment layout ----------------
// Bmat[k][n]: k<8960 -> coef[k][n]*spline_scale[k/35][n]; k>=8960 -> base_scale[k-8960][n].
// Fragment element index e = ((kc*4 + nt)*32 + lane)*16 + kk,
// lane<16 holds K rows kc*32+{0..7,16..23} of column n=(lane&15)+nt*16; lane>=16 holds K rows kc*32+{8..15,24..31}.
__global__ __launch_bounds__(256)
void kan_pack_b(const float* __restrict__ coef,
                const float* __restrict__ spline_scale,
                const float* __restrict__ base_scale,
                _Float16* __restrict__ wsB) {
  int e = blockIdx.x * blockDim.x + threadIdx.x;
  if (e >= BFRAG_ELEMS) return;
  int kk = e & 15;
  int l  = (e >> 4) & 31;
  int nt = (e >> 9) & 3;
  int kc = e >> 11;
  int k = kc * 32 + ((l < 16) ? 0 : 8) + (kk & 7) + ((kk & 8) << 1);
  int n = (l & 15) + nt * 16;
  float v;
  if (k < KSPLINE) {
    int d = k / NCOEF;
    v = coef[k * OUT_DIM + n] * spline_scale[d * OUT_DIM + n];
  } else {
    int d = k - KSPLINE;
    v = base_scale[d * OUT_DIM + n];
  }
  wsB[e] = (_Float16)v;
}

// ---------------- kernel 3: scatter-A + pipelined WMMA GEMM ----------------
// Block = 16 batch rows, 128 threads = 4 waves; wave w computes N-tile w.
// LDS: 48KB A panel (16 x 1536 f16) + 16KB cached xn (16 x 256 f32).
__global__ __launch_bounds__(128)
void kan_wmma_main(const float* __restrict__ x,
                   const float* __restrict__ mean,
                   const float* __restrict__ scale,
                   const _Float16* __restrict__ wsB,
                   float* __restrict__ out) {
  __shared__ _Float16 Alds[16 * SCK];     // 49152 bytes
  __shared__ float    Xn[16 * IN_DIM];    // 16384 bytes
  const int tid  = threadIdx.x;
  const int lane = tid & 31;
  const int wave = tid >> 5;              // N-tile index 0..3
  const int rowBase = blockIdx.x * 16;

  // ---- phase 0: normalize x tile once into LDS ----
  for (int i = tid; i < 16 * IN_DIM; i += 128) {
    int r = i >> 8, d = i & (IN_DIM - 1);
    Xn[i] = (x[(size_t)(rowBase + r) * IN_DIM + d] - mean[d]) * scale[d];
  }
  __syncthreads();

  v8f acc = {};

  for (int sc = 0; sc < NSC; ++sc) {
    const int scLo = sc * SCK;
    const int scHi = scLo + SCK;

    // ---- zero the A panel (b128 stores) ----
    uint4 zz; zz.x = 0u; zz.y = 0u; zz.z = 0u; zz.w = 0u;
    uint4* Az = (uint4*)Alds;
    #pragma unroll
    for (int i = 0; i < (16 * SCK * 2 / 16) / 128; ++i)   // 24 per thread
      Az[tid + 128 * i] = zz;
    __syncthreads();

    // ---- scatter spline basis: k = d*35 + idx + j ----
    {
      int dLo = (scLo >= (NCOEF - 1)) ? (scLo - (NCOEF - 1)) / NCOEF : 0;
      int dHi = (scHi - 1) / NCOEF; if (dHi > IN_DIM - 1) dHi = IN_DIM - 1;
      int nd = dHi - dLo + 1;
      for (int t = tid; t < 16 * nd; t += 128) {
        int r = t / nd;
        int d = dLo + (t - r * nd);
        float xn = Xn[(r << 8) + d];
        float xc = fminf(fmaxf(xn, -1.f), 1.f);
        float u  = (xc + 1.f) * 16.f;          // xs / gap, gap = 1/16
        float fu = floorf(u);
        int idx = (int)fu; if (idx > N_INT - 1) idx = N_INT - 1;
        float tt = u - fu;                     // == mod(xs,gap)/gap (x>=0)
        float t2 = tt * tt, t3 = t2 * tt;
        float omt = 1.f - tt;
        float b0 = omt * omt * omt * (1.f / 6.f);
        float b1 = (3.f * t3 - 6.f * t2 + 4.f) * (1.f / 6.f);
        float b2 = (-3.f * t3 + 3.f * t2 + 3.f * tt + 1.f) * (1.f / 6.f);
        float b3 = t3 * (1.f / 6.f);
        float bs[4] = { b0, b1, b2, b3 };
        int k0 = d * NCOEF + idx;
        #pragma unroll
        for (int j = 0; j < 4; ++j) {
          int k = k0 + j;
          if (k >= scLo && k < scHi)
            Alds[r * SCK + (k - scLo)] = (_Float16)bs[j];
        }
      }
    }
    // ---- scatter fused base term: k = 8960 + d holds silu(xn) ----
    {
      int a = scLo - KSPLINE; if (a < 0) a = 0;
      int bnd = scHi - KSPLINE; if (bnd > IN_DIM) bnd = IN_DIM;
      int ndb = bnd - a;
      if (ndb > 0) {
        for (int t = tid; t < 16 * ndb; t += 128) {
          int r = t / ndb;
          int d = a + (t - r * ndb);
          float xn = Xn[(r << 8) + d];
          float sl = xn / (1.f + __expf(-xn));
          Alds[r * SCK + (KSPLINE + d - scLo)] = (_Float16)sl;
        }
      }
    }
    __syncthreads();

    // ---- software-pipelined WMMA over 48 chunks ----
    // A frag layout: lane<16 -> row lane, K 0..7 / 16..23; lane>=16 -> row lane-16, K 8..15 / 24..31.
    const int r16 = lane & 15;
    const int hi8 = (lane >> 4) * 8;
    const _Float16* Abase = Alds + r16 * SCK + hi8;
    const _Float16* Bbase = wsB + ((((size_t)sc * CPS) * NTILES + wave) * 32 + lane) * 16;

    union F { v16h v; uint4 q[2]; };
    F ac, bc;
    ac.q[0] = *(const uint4*)(Abase);                 // ds_load_b128
    ac.q[1] = *(const uint4*)(Abase + 16);
    bc.q[0] = *(const uint4*)(Bbase);                 // global_load_b128
    bc.q[1] = *(const uint4*)(Bbase + 8);

    #pragma unroll 4
    for (int c = 0; c < CPS - 1; ++c) {
      // issue next chunk's fragment loads before consuming current ones
      F an, bn;
      const _Float16* ap = Abase + (c + 1) * KCHUNK;
      const _Float16* bq = Bbase + (size_t)(c + 1) * BCSTRIDE;
      an.q[0] = *(const uint4*)(ap);
      an.q[1] = *(const uint4*)(ap + 16);
      bn.q[0] = *(const uint4*)(bq);
      bn.q[1] = *(const uint4*)(bq + 8);
      if (sc + 1 < NSC)   // warm L2/L0 for the next super-chunk's B panel
        __builtin_prefetch(Bbase + (size_t)(CPS + c) * BCSTRIDE, 0, 0);
      acc = __builtin_amdgcn_wmma_f32_16x16x32_f16(
          false, ac.v, false, bc.v, (short)0, acc, false, false);
      ac = an; bc = bn;
    }
    acc = __builtin_amdgcn_wmma_f32_16x16x32_f16(
        false, ac.v, false, bc.v, (short)0, acc, false, false);
    __syncthreads();
  }

  // ---- write D: VGPR i -> M=i (lanes 0-15), M=8+i (lanes 16-31) ----
  const int col  = wave * 16 + (lane & 15);
  const int rOff = (lane >> 4) * 8;
  #pragma unroll
  for (int i = 0; i < 8; ++i)
    out[(size_t)(rowBase + rOff + i) * OUT_DIM + col] = acc[i];
}

// ---------------- launcher ----------------
extern "C" void kernel_launch(void* const* d_in, const int* in_sizes, int n_in,
                              void* d_out, int out_size, void* d_ws, size_t ws_size,
                              hipStream_t stream) {
  const float* x            = (const float*)d_in[0];   // (8192,256)
  const float* coef         = (const float*)d_in[1];   // (256,35,64)
  const float* spline_scale = (const float*)d_in[2];   // (256,64)
  const float* base_scale   = (const float*)d_in[3];   // (256,64)
  float* out = (float*)d_out;                          // (8192,64) f32

  // workspace map: [0,1024) mean f32, [1024,2048) scale f32, [2048, +1.125MB) B frags f16
  float* wsMean  = (float*)d_ws;
  float* wsScale = wsMean + IN_DIM;
  _Float16* wsB  = (_Float16*)((char*)d_ws + 2048);

  kan_bn_stats<<<IN_DIM, 256, 0, stream>>>(x, wsMean, wsScale);
  kan_pack_b<<<(BFRAG_ELEMS + 255) / 256, 256, 0, stream>>>(coef, spline_scale, base_scale, wsB);
  kan_wmma_main<<<BATCH / 16, 128, 0, stream>>>(x, wsMean, wsScale, wsB, out);
}
